// CRec_89026082111511
// MI455X (gfx1250) — matrix operand order repaired and checked by reading, not compiled, not run on hardware
//
#include <hip/hip_runtime.h>
#include <math.h>

// ---------------------------------------------------------------------------
// DIN-style attention recommender forward loss, MI455X (gfx1250, wave32).
//
// Algebraic fusion: logits[s] = h[s]·(Wk^T q),  agg = Wv·(Σ score·h) + bv.
// -> k/v projections eliminated (25x FLOP cut); kernel is L2-gather bound.
// WMMA (v_wmma_f32_16x16x32_bf16) used for per-tile logits (qk replicated
// across B columns) and for the batched Wv/Wp tail GEMM.
// ---------------------------------------------------------------------------

#define DIMN   64
#define SEQ    200
#define BATCH  8192
#define NEGC   (-4294967296.0f)   // -2^32, exact reference mask constant
#define WPB    8                  // waves per block in attention kernel

typedef __attribute__((ext_vector_type(16))) __bf16          v16bf;
typedef __attribute__((ext_vector_type(16))) unsigned short  v16us;
typedef __attribute__((ext_vector_type(8)))  float           v8f;

__device__ __forceinline__ unsigned short f2bf(float x) {
  unsigned int u = __float_as_uint(x);
  unsigned int r = (u + 0x7fffu + ((u >> 16) & 1u)) >> 16;   // RNE
  return (unsigned short)r;
}
__device__ __forceinline__ unsigned int pack_bf2(float lo, float hi) {
  return (unsigned int)f2bf(lo) | ((unsigned int)f2bf(hi) << 16);
}
__device__ __forceinline__ float bfsel(unsigned int pair, int odd) {
  return __uint_as_float(odd ? (pair & 0xffff0000u) : (pair << 16));
}

__device__ __forceinline__ v8f wmma_bf16(v16us a, v16us b, v8f c) {
  return __builtin_amdgcn_wmma_f32_16x16x32_bf16(
      false, __builtin_bit_cast(v16bf, a),
      false, __builtin_bit_cast(v16bf, b),
      (short)0, c, false, false);
}

// ===========================================================================
// Kernel 1: one wave32 per batch row. Gathers h tiles (16 tokens x 64 dims)
// as bf16 into LDS, WMMA logits against replicated qk, online softmax,
// accumulates hbar = sum(score * h). Writes hbar to workspace.
// ===========================================================================
__global__ void __launch_bounds__(32 * WPB)
crec_attn(const int* __restrict__ hist, const int* __restrict__ cand,
          const float* __restrict__ emb,
          const float* __restrict__ Wq, const float* __restrict__ bq,
          const float* __restrict__ Wk, const float* __restrict__ bk,
          float* __restrict__ hb_out) {
  __shared__ unsigned int hp[WPB][16][33];  // h tile: 16 tokens x 32 bf16-pairs (+pad)
  __shared__ float cb[WPB][DIMN];           // candidate embedding
  __shared__ float qb[WPB][DIMN];           // q
  __shared__ float qkb[WPB][DIMN];          // qk = Wk^T q
  __shared__ float mk[WPB][16];             // token mask (hist != 0)
  __shared__ float wb[WPB][16];             // softmax weights per tile

  const int lane = threadIdx.x & 31;
  const int wv   = threadIdx.x >> 5;
  const int b    = blockIdx.x * WPB + wv;   // grid = BATCH / WPB
  const int nlo  = lane & 15;
  const int half = lane >> 4;

  // ---- candidate embedding (padding_idx=0 -> zero row) ----
  {
    const int cidx = cand[b];
    float c0 = 0.0f, c1 = 0.0f;
    if (cidx != 0) {
      c0 = emb[cidx * DIMN + lane];
      c1 = emb[cidx * DIMN + lane + 32];
    }
    cb[wv][lane]      = c0;
    cb[wv][lane + 32] = c1;
  }
  __syncthreads();

  // ---- q[n] = sum_d c[d]*Wq[n,d] + bq[n]; lane owns n=lane, lane+32 ----
  float q0 = bq[lane], q1 = bq[lane + 32];
  #pragma unroll 4
  for (int d = 0; d < DIMN; d += 4) {
    const float4 w0 = *(const float4*)(Wq + lane * DIMN + d);
    const float4 w1 = *(const float4*)(Wq + (lane + 32) * DIMN + d);
    const float4 cc = *(const float4*)(&cb[wv][d]);
    q0 += w0.x * cc.x + w0.y * cc.y + w0.z * cc.z + w0.w * cc.w;
    q1 += w1.x * cc.x + w1.y * cc.y + w1.z * cc.z + w1.w * cc.w;
  }
  qb[wv][lane]      = q0;
  qb[wv][lane + 32] = q1;

  // ---- lconst = q . bk (bias of k folds into a constant logit offset) ----
  float lpart = q0 * bk[lane] + q1 * bk[lane + 32];
  lpart += __shfl_xor(lpart, 1);
  lpart += __shfl_xor(lpart, 2);
  lpart += __shfl_xor(lpart, 4);
  lpart += __shfl_xor(lpart, 8);
  lpart += __shfl_xor(lpart, 16);
  const float lconst = lpart;
  __syncthreads();

  // ---- qk[d] = sum_n q[n]*Wk[n,d]; lane owns d=lane, lane+32 ----
  {
    float a0 = 0.0f, a1 = 0.0f;
    #pragma unroll 4
    for (int n = 0; n < DIMN; ++n) {
      const float qn = qb[wv][n];
      a0 += qn * Wk[n * DIMN + lane];
      a1 += qn * Wk[n * DIMN + lane + 32];
    }
    qkb[wv][lane]      = a0;
    qkb[wv][lane + 32] = a1;
  }
  __syncthreads();

  // ---- B fragments: qk replicated into all 16 columns (ISA B layout:
  //      lane: n=lane&15, half=lane>>4; VGPR j packs K = 16*(j>>2) + 8*half + 2*(j&3) + t) ----
  v16us bu0, bu1;
  #pragma unroll
  for (int j = 0; j < 8; ++j) {
    const int kk = ((j >> 2) << 4) + ((j & 3) << 1) + (half << 3);
    bu0[2 * j]     = f2bf(qkb[wv][kk]);
    bu0[2 * j + 1] = f2bf(qkb[wv][kk + 1]);
    bu1[2 * j]     = f2bf(qkb[wv][32 + kk]);
    bu1[2 * j + 1] = f2bf(qkb[wv][33 + kk]);
  }

  // ---- online-softmax tile loop ----
  float runM = -__builtin_inff();
  float Z = 0.0f, hb0 = 0.0f, hb1 = 0.0f;
  const int tok = lane >> 1;          // gather: 2 lanes per token
  const int dh  = (lane & 1) << 5;    // 32 dims each

  for (int s0 = 0; s0 < SEQ; s0 += 16) {
    const int s = s0 + tok;
    int idx = 0;
    if (s < SEQ) idx = hist[b * SEQ + s];
    if ((lane & 1) == 0) mk[wv][tok] = (idx != 0) ? 1.0f : 0.0f;
    if (s0 + 16 < SEQ) __builtin_prefetch(&hist[b * SEQ + s0 + 16 + tok], 0, 0);

    const float* hsrc = emb + idx * DIMN + dh;
    #pragma unroll
    for (int d = 0; d < 32; d += 4) {
      float4 v = make_float4(0.f, 0.f, 0.f, 0.f);
      if (idx != 0) v = *(const float4*)(hsrc + d);
      hp[wv][tok][(dh + d) >> 1]     = pack_bf2(v.x, v.y);
      hp[wv][tok][(dh + d + 2) >> 1] = pack_bf2(v.z, v.w);
    }
    __syncthreads();

    // A fragments (ISA 16-bit A layout: kk = (j>>2)*16 + (j&3)*2 + 8*half + t)
    v16us au0, au1;
    #pragma unroll
    for (int j = 0; j < 8; ++j) {
      const int pi = (((j >> 2) << 4) + ((j & 3) << 1) + (half << 3)) >> 1;
      const unsigned int u0 = hp[wv][nlo][pi];
      const unsigned int u1 = hp[wv][nlo][16 + pi];
      au0[2 * j]     = (unsigned short)(u0 & 0xffffu);
      au0[2 * j + 1] = (unsigned short)(u0 >> 16);
      au1[2 * j]     = (unsigned short)(u1 & 0xffffu);
      au1[2 * j + 1] = (unsigned short)(u1 >> 16);
    }

    v8f lf = {};
    lf = wmma_bf16(au0, bu0, lf);   // K = 0..31
    lf = wmma_bf16(au1, bu1, lf);   // K = 32..63

    // logits (replicated across n-lanes; lane holds m = r + 8*half)
    float lg[8];
    float tmax = -__builtin_inff();
    #pragma unroll
    for (int r = 0; r < 8; ++r) {
      const int m = r + (half << 3);
      float x = lf[r] + lconst;
      const float mv = mk[wv][m];
      x = x * mv + x * (1.0f - mv) * NEGC;      // exact reference mask formula
      if (s0 + m >= SEQ) x = -__builtin_inff(); // padded tokens: weight 0
      lg[r] = x;
      tmax = fmaxf(tmax, x);
    }
    tmax = fmaxf(tmax, __shfl_xor(tmax, 16));

    const float nM = fmaxf(runM, tmax);
    const float rs = __expf(runM - nM);
    Z *= rs; hb0 *= rs; hb1 *= rs;

    float zloc = 0.0f, wr[8];
    #pragma unroll
    for (int r = 0; r < 8; ++r) { wr[r] = __expf(lg[r] - nM); zloc += wr[r]; }
    Z += zloc + __shfl_xor(zloc, 16);
    runM = nM;

    if (nlo == 0) {
      #pragma unroll
      for (int r = 0; r < 8; ++r) wb[wv][r + (half << 3)] = wr[r];
    }
    __syncthreads();

    // hbar accumulation: lane owns dims lane, lane+32
    #pragma unroll
    for (int m = 0; m < 16; ++m) {
      const float wm = wb[wv][m];
      const unsigned int u0 = hp[wv][m][lane >> 1];
      const unsigned int u1 = hp[wv][m][(lane + 32) >> 1];
      hb0 += wm * bfsel(u0, lane & 1);
      hb1 += wm * bfsel(u1, lane & 1);
    }
    __syncthreads();
  }

  const float invZ = 1.0f / Z;
  hb_out[b * DIMN + lane]      = hb0 * invZ;
  hb_out[b * DIMN + lane + 32] = hb1 * invZ;
}

// ===========================================================================
// Kernel 2: batched tail. One wave per 16-batch tile:
//   agg = hbar @ Wv^T + bv ; p = agg @ Wp^T + bp ; out = p @ Wc^T + bc
//   loss += sum over tile of (logsumexp(out) - out[label]) / BATCH
// ===========================================================================
__global__ void __launch_bounds__(32)
crec_tail(const float* __restrict__ hb, const int* __restrict__ label,
          const float* __restrict__ Wv, const float* __restrict__ bv,
          const float* __restrict__ Wp, const float* __restrict__ bp,
          const float* __restrict__ Wc, const float* __restrict__ bc,
          float* __restrict__ out) {
  __shared__ unsigned int ap[16][33];
  __shared__ float tb[16][DIMN];

  const int lane = threadIdx.x;
  const int b0   = blockIdx.x << 4;
  const int nlo  = lane & 15;
  const int half = lane >> 4;

  // stage hbar tile (16 batches x 64) as bf16 pairs
  {
    const int row = lane >> 1, dh = (lane & 1) << 5;
    #pragma unroll
    for (int d = 0; d < 32; d += 4) {
      const float4 v = *(const float4*)(hb + (b0 + row) * DIMN + dh + d);
      ap[row][(dh + d) >> 1]     = pack_bf2(v.x, v.y);
      ap[row][(dh + d + 2) >> 1] = pack_bf2(v.z, v.w);
    }
  }
  __syncthreads();

  // A fragments for GEMM1
  v16us au0, au1;
  #pragma unroll
  for (int j = 0; j < 8; ++j) {
    const int pi = (((j >> 2) << 4) + ((j & 3) << 1) + (half << 3)) >> 1;
    const unsigned int u0 = ap[nlo][pi];
    const unsigned int u1 = ap[nlo][16 + pi];
    au0[2 * j]     = (unsigned short)(u0 & 0xffffu);
    au0[2 * j + 1] = (unsigned short)(u0 >> 16);
    au1[2 * j]     = (unsigned short)(u1 & 0xffffu);
    au1[2 * j + 1] = (unsigned short)(u1 >> 16);
  }

  // GEMM1: agg = hbar @ Wv^T + bv  (N tiled 4x16)
  #pragma unroll
  for (int nt = 0; nt < 4; ++nt) {
    const int rowW = (nt << 4) + nlo;
    v16us bu0, bu1;
    #pragma unroll
    for (int j = 0; j < 8; ++j) {
      const int kk = ((j >> 2) << 4) + ((j & 3) << 1) + (half << 3);
      bu0[2 * j]     = f2bf(Wv[rowW * DIMN + kk]);
      bu0[2 * j + 1] = f2bf(Wv[rowW * DIMN + kk + 1]);
      bu1[2 * j]     = f2bf(Wv[rowW * DIMN + 32 + kk]);
      bu1[2 * j + 1] = f2bf(Wv[rowW * DIMN + 33 + kk]);
    }
    v8f f = {};
    f = wmma_bf16(au0, bu0, f);
    f = wmma_bf16(au1, bu1, f);
    const float bias = bv[rowW];
    #pragma unroll
    for (int r = 0; r < 8; ++r) tb[r + (half << 3)][rowW] = f[r] + bias;
  }
  __syncthreads();

  // A' fragments from agg (f32 -> bf16)
  v16us cu0, cu1;
  #pragma unroll
  for (int j = 0; j < 8; ++j) {
    const int kk = ((j >> 2) << 4) + ((j & 3) << 1) + (half << 3);
    cu0[2 * j]     = f2bf(tb[nlo][kk]);
    cu0[2 * j + 1] = f2bf(tb[nlo][kk + 1]);
    cu1[2 * j]     = f2bf(tb[nlo][32 + kk]);
    cu1[2 * j + 1] = f2bf(tb[nlo][33 + kk]);
  }

  // GEMM2 + fold Wc: per-lane partials over this lane's n columns
  float po0[8], po1[8];
  #pragma unroll
  for (int r = 0; r < 8; ++r) { po0[r] = 0.0f; po1[r] = 0.0f; }

  #pragma unroll
  for (int nt = 0; nt < 4; ++nt) {
    const int rowW = (nt << 4) + nlo;
    v16us bu0, bu1;
    #pragma unroll
    for (int j = 0; j < 8; ++j) {
      const int kk = ((j >> 2) << 4) + ((j & 3) << 1) + (half << 3);
      bu0[2 * j]     = f2bf(Wp[rowW * DIMN + kk]);
      bu0[2 * j + 1] = f2bf(Wp[rowW * DIMN + kk + 1]);
      bu1[2 * j]     = f2bf(Wp[rowW * DIMN + 32 + kk]);
      bu1[2 * j + 1] = f2bf(Wp[rowW * DIMN + 33 + kk]);
    }
    v8f g = {};
    g = wmma_bf16(cu0, bu0, g);
    g = wmma_bf16(cu1, bu1, g);
    const float bpv = bp[rowW];
    const float wc0 = Wc[rowW];
    const float wc1 = Wc[DIMN + rowW];
    #pragma unroll
    for (int r = 0; r < 8; ++r) {
      const float pv = g[r] + bpv;
      po0[r] += pv * wc0;
      po1[r] += pv * wc1;
    }
  }

  // reduce over the 16 n-lanes
  #pragma unroll
  for (int r = 0; r < 8; ++r) {
    #pragma unroll
    for (int off = 1; off < 16; off <<= 1) {
      po0[r] += __shfl_xor(po0[r], off);
      po1[r] += __shfl_xor(po1[r], off);
    }
  }

  if (nlo == 0) {
    float acc = 0.0f;
    #pragma unroll
    for (int r = 0; r < 8; ++r) {
      const int bb = b0 + r + (half << 3);
      const float o0 = po0[r] + bc[0];
      const float o1 = po1[r] + bc[1];
      const float mx = fmaxf(o0, o1);
      const float lse = mx + __logf(__expf(o0 - mx) + __expf(o1 - mx));
      acc += lse - (label[bb] ? o1 : o0);
    }
    atomicAdd(out, acc * (1.0f / (float)BATCH));
  }
}

__global__ void crec_zero(float* __restrict__ o) {
  if (threadIdx.x == 0 && blockIdx.x == 0) o[0] = 0.0f;
}

// ===========================================================================
extern "C" void kernel_launch(void* const* d_in, const int* in_sizes, int n_in,
                              void* d_out, int out_size, void* d_ws, size_t ws_size,
                              hipStream_t stream) {
  (void)in_sizes; (void)n_in; (void)out_size; (void)ws_size;
  const int*   hist  = (const int*)d_in[0];
  const int*   cand  = (const int*)d_in[1];
  const int*   label = (const int*)d_in[2];
  const float* emb   = (const float*)d_in[3];
  const float* Wq    = (const float*)d_in[4];
  const float* bq    = (const float*)d_in[5];
  const float* Wk    = (const float*)d_in[6];
  const float* bk    = (const float*)d_in[7];
  const float* Wv    = (const float*)d_in[8];
  const float* bv    = (const float*)d_in[9];
  const float* Wp    = (const float*)d_in[10];
  const float* bp    = (const float*)d_in[11];
  const float* Wc    = (const float*)d_in[12];
  const float* bc    = (const float*)d_in[13];
  float* loss = (float*)d_out;
  float* hbws = (float*)d_ws;   // BATCH * 64 floats = 2 MB scratch

  crec_zero<<<1, 1, 0, stream>>>(loss);
  crec_attn<<<BATCH / WPB, 32 * WPB, 0, stream>>>(hist, cand, emb,
                                                  Wq, bq, Wk, bk, hbws);
  crec_tail<<<BATCH / 16, 32, 0, stream>>>(hbws, label, Wv, bv, Wp, bp,
                                           Wc, bc, loss);
}
